// MambaBlock_26517128086026
// MI455X (gfx1250) — compile-verified, matrix operand order checked by reading, not run on hardware
//
#include <hip/hip_runtime.h>

// ---------------------------------------------------------------------------
// Mamba block forward for MI455X (gfx1250, wave32, WMMA).
//   1. fp32->bf16 converts (weights + x)
//   2. GEMM1 (WMMA bf16, 32x64/wave): x @ W_in.T -> x_and_res fp32
//   3. causal depthwise conv (K=4) + bias + SiLU -> x_m bf16
//   4. GEMM2 (WMMA, 16x16/wave, N=96): x_m @ W_x.T -> x_dbl fp32
//   5. slice dlt -> bf16
//   6. GEMM3 (WMMA 32x64/wave): dlt @ W_dt.T, +b_dt, softplus -> delta fp32
//   7. selective scan (async B/C staging to LDS, states in regs) fused with
//      +D*x skip, *silu(res) gate, bf16 downconvert -> y bf16
//   8. GEMM4 (WMMA 32x64/wave): y @ W_out.T -> out fp32
// ---------------------------------------------------------------------------

#define MD_MODEL 1024
#define MD_STATE 16
#define MD_CONV  4
#define MD_INNER 2048
#define MDT_RANK 64
#define MB       2
#define ML       2048
#define NTOK     (MB * ML)                    // 4096 tokens
#define XDBL_C   (MDT_RANK + 2 * MD_STATE)    // 96

typedef __attribute__((ext_vector_type(16))) __bf16 v16bf_t;
typedef __attribute__((ext_vector_type(8)))  float  v8f_t;

__device__ __forceinline__ unsigned short f32_bf16(float f) {
    unsigned int u = __float_as_uint(f);
    u += 0x7FFFu + ((u >> 16) & 1u);           // round-to-nearest-even
    return (unsigned short)(u >> 16);
}
__device__ __forceinline__ float bf16_f32(unsigned short h) {
    return __uint_as_float(((unsigned int)h) << 16);
}

// -------------------- fp32 -> bf16 convert (grid-stride) -------------------
__global__ void k_cvt_bf16(const float* __restrict__ s,
                           unsigned short* __restrict__ d, int n) {
    for (int i = blockIdx.x * blockDim.x + threadIdx.x; i < n;
         i += gridDim.x * blockDim.x)
        d[i] = f32_bf16(s[i]);
}

// strided slice convert: dst[r,c] = bf16(src[r*src_ld + c]) for c < cols
__global__ void k_slice_bf16(const float* __restrict__ src,
                             unsigned short* __restrict__ dst,
                             int rows, int src_ld, int cols) {
    int i = blockIdx.x * blockDim.x + threadIdx.x;
    if (i >= rows * cols) return;
    int r = i / cols, c = i - r * cols;
    dst[i] = f32_bf16(src[(size_t)r * src_ld + c]);
}

// -------------------- WMMA frag load ---------------------------------------
// A-frag (16x32 bf16): lane(0-15)=row M, holds K {k0..k0+7, k0+16..k0+23};
// lanes 16-31 hold {k0+8..k0+15, k0+24..k0+31}. Two contiguous b128 loads.
// B-frag mirrored (lane = N index, same K packing) = row of [N,K] weights.
__device__ __forceinline__ v16bf_t ld_frag16(const unsigned short* p) {
    uint4 q[2];
    q[0] = *reinterpret_cast<const uint4*>(p);        // K = k .. k+7
    q[1] = *reinterpret_cast<const uint4*>(p + 16);   // K = k+16 .. k+23
    v16bf_t r;
    __builtin_memcpy(&r, q, sizeof(r));
    return r;
}

// -------------------- WMMA GEMM, 32x64 output block per wave ---------------
// C[M,N] = A[M,K] * W[N,K]^T ; Mtiles%2==0, Ntiles%4==0.
// Per k-step: 2 A-frags + 4 B-frags (all in distinct registers, so the
// following loads have no WAR hazard against the WMMAs) feed 8 independent
// WMMA accumulator chains.
// epilogue: 0 = plain fp32 store, 1 = +bias[col] then softplus
__global__ void k_gemm_wmma_2x4(const unsigned short* __restrict__ A, int lda,
                                const unsigned short* __restrict__ W, int ldw,
                                float* __restrict__ C, int ldc,
                                int Mt2, int Nt4, int K,
                                const float* __restrict__ bias, int epilogue) {
    int wave = blockIdx.x * (blockDim.x >> 5) + (threadIdx.x >> 5);
    if (wave >= Mt2 * Nt4) return;         // wave-uniform: EXEC stays all-1s
    int lane = threadIdx.x & 31;
    int tm2 = wave % Mt2;                  // 32-row block
    int tn4 = wave / Mt2;                  // 64-col block (waves share W tiles)
    int half = lane >> 4;
    int r    = lane & 15;

    const unsigned short* pa0 = A + (size_t)(tm2 * 32 + r) * lda + half * 8;
    const unsigned short* pa1 = pa0 + (size_t)16 * lda;
    const unsigned short* pw0 = W + (size_t)(tn4 * 64 + r) * ldw + half * 8;
    const unsigned short* pw1 = pw0 + (size_t)16 * ldw;
    const unsigned short* pw2 = pw0 + (size_t)32 * ldw;
    const unsigned short* pw3 = pw0 + (size_t)48 * ldw;

    v8f_t acc[2][4];
#pragma unroll
    for (int i2 = 0; i2 < 2; i2++)
#pragma unroll
        for (int j = 0; j < 4; j++) acc[i2][j] = (v8f_t){};

    for (int k0 = 0; k0 < K; k0 += 32) {
        // WGP-scope speculative prefetch one cacheline (128B) ahead on all
        // six streams (drop-on-translate-fail safe at the tail)
        __builtin_prefetch(pa0 + k0 + 64, 0, 3);
        __builtin_prefetch(pa1 + k0 + 64, 0, 3);
        __builtin_prefetch(pw0 + k0 + 64, 0, 3);
        __builtin_prefetch(pw1 + k0 + 64, 0, 3);
        __builtin_prefetch(pw2 + k0 + 64, 0, 3);
        __builtin_prefetch(pw3 + k0 + 64, 0, 3);

        v16bf_t a0 = ld_frag16(pa0 + k0);
        v16bf_t a1 = ld_frag16(pa1 + k0);
        v16bf_t bf[4];
        bf[0] = ld_frag16(pw0 + k0);
        bf[1] = ld_frag16(pw1 + k0);
        bf[2] = ld_frag16(pw2 + k0);
        bf[3] = ld_frag16(pw3 + k0);
#pragma unroll
        for (int j = 0; j < 4; j++) {
            acc[0][j] = __builtin_amdgcn_wmma_f32_16x16x32_bf16(
                false, a0, false, bf[j], (short)0, acc[0][j], false, false);
            acc[1][j] = __builtin_amdgcn_wmma_f32_16x16x32_bf16(
                false, a1, false, bf[j], (short)0, acc[1][j], false, false);
        }
    }

    // C/D layout: element i -> row base + i + 8*half, col base + r
#pragma unroll
    for (int i2 = 0; i2 < 2; i2++) {
#pragma unroll
        for (int j = 0; j < 4; j++) {
            int col = tn4 * 64 + j * 16 + r;
            float* cp =
                C + (size_t)(tm2 * 32 + i2 * 16 + half * 8) * ldc + col;
            float bv = (epilogue == 1) ? bias[col] : 0.f;
#pragma unroll
            for (int i = 0; i < 8; i++) {
                float v = acc[i2][j][i];
                if (epilogue == 1) {
                    v += bv;
                    v = (v > 20.f) ? v : log1pf(__expf(v));  // softplus
                }
                cp[(size_t)i * ldc] = v;
            }
        }
    }
}

// -------------------- WMMA GEMM, one 16x16 tile per wave (small N) ---------
__global__ void k_gemm_wmma(const unsigned short* __restrict__ A, int lda,
                            const unsigned short* __restrict__ W, int ldw,
                            float* __restrict__ C, int ldc,
                            int Mtiles, int Ntiles, int K) {
    int wave = blockIdx.x * (blockDim.x >> 5) + (threadIdx.x >> 5);
    if (wave >= Mtiles * Ntiles) return;
    int lane = threadIdx.x & 31;
    int tm = wave % Mtiles;
    int tn = wave / Mtiles;
    int half = lane >> 4;
    int r    = lane & 15;

    const unsigned short* pa = A + (size_t)(tm * 16 + r) * lda + half * 8;
    const unsigned short* pw = W + (size_t)(tn * 16 + r) * ldw + half * 8;

    v8f_t acc = {};
    for (int k0 = 0; k0 < K; k0 += 32) {
        v16bf_t af = ld_frag16(pa + k0);
        v16bf_t wf = ld_frag16(pw + k0);
        acc = __builtin_amdgcn_wmma_f32_16x16x32_bf16(
            false, af, false, wf, (short)0, acc, false, false);
    }
    int col = tn * 16 + r;
    float* cp = C + (size_t)(tm * 16 + half * 8) * ldc + col;
#pragma unroll
    for (int i = 0; i < 8; i++) cp[(size_t)i * ldc] = acc[i];
}

// -------------------- causal depthwise conv1d (K=4) + bias + SiLU ----------
__global__ void k_conv_silu(const float* __restrict__ xr,
                            const float* __restrict__ conv_w,
                            const float* __restrict__ conv_b,
                            unsigned short* __restrict__ xm_bf) {
    int d = blockIdx.x * blockDim.x + threadIdx.x;   // channel (coalesced)
    int l = blockIdx.y;
    int b = blockIdx.z;
    const float* base = xr + ((size_t)b * ML + l) * (2 * MD_INNER) + d;
    const float* w = conv_w + d * MD_CONV;
    float acc = conv_b[d];
#pragma unroll
    for (int j = 0; j < MD_CONV; j++) {
        int li = l - (MD_CONV - 1) + j;              // left-pad causal
        if (li >= 0)
            acc += w[j] * base[(long)(j - (MD_CONV - 1)) * (2 * MD_INNER)];
    }
    float s = acc / (1.f + __expf(-acc));            // SiLU
    xm_bf[((size_t)b * ML + l) * MD_INNER + d] = f32_bf16(s);
}

// -------------------- selective scan + skip + gate (fused) -----------------
// one thread per (b,d) channel, 16 states in registers, serial over L.
// Per-step B_t/C_t (32 floats) staged to LDS with async global->LDS copies
// (ASYNCcnt-tracked, no VGPR return path), lanes 0..7 move 16B each.
__global__ void k_scan_gate(const float* __restrict__ delta,
                            const unsigned short* __restrict__ xm_bf,
                            const float* __restrict__ xdbl,
                            const float* __restrict__ xr,
                            const float* __restrict__ A_log,
                            const float* __restrict__ D_par,
                            unsigned short* __restrict__ y_bf) {
    __shared__ __align__(16) float sBC[2 * MD_STATE];  // [0..15]=B, [16..31]=C
    int d = blockIdx.x * blockDim.x + threadIdx.x;
    int b = blockIdx.y;

    float Aa[MD_STATE], h[MD_STATE];
#pragma unroll
    for (int n = 0; n < MD_STATE; n++) {
        Aa[n] = -__expf(A_log[d * MD_STATE + n]);    // A = -exp(A_log)
        h[n] = 0.f;
    }
    float Dp = D_par[d];

    for (int l = 0; l < ML; l++) {
        size_t tok = (size_t)b * ML + l;
        // async copy 32 floats (B_t|C_t) from x_dbl cols 64..95 into LDS
        if (threadIdx.x < 8) {
            unsigned lds_off = (unsigned)(size_t)(&sBC[threadIdx.x * 4]);
            const float* g = xdbl + tok * XDBL_C + MDT_RANK + threadIdx.x * 4;
            asm volatile("global_load_async_to_lds_b128 %0, %1, off"
                         :: "v"(lds_off), "v"(g) : "memory");
        }
        asm volatile("s_wait_asynccnt 0x0" ::: "memory");
        __syncthreads();

        size_t idx = tok * MD_INNER + d;
        float dl = delta[idx];
        float xv = bf16_f32(xm_bf[idx]);
        float dx = dl * xv;
        float y = 0.f;
#pragma unroll
        for (int n = 0; n < MD_STATE; n++) {
            h[n] = __expf(dl * Aa[n]) * h[n] + dx * sBC[n];
            y += h[n] * sBC[MD_STATE + n];
        }
        y += Dp * xv;                                       // skip term
        float rg = xr[tok * (2 * MD_INNER) + MD_INNER + d]; // res
        y *= rg / (1.f + __expf(-rg));                      // * silu(res)
        y_bf[idx] = f32_bf16(y);
        __syncthreads();
    }
}

// ---------------------------------------------------------------------------
extern "C" void kernel_launch(void* const* d_in, const int* in_sizes, int n_in,
                              void* d_out, int out_size, void* d_ws, size_t ws_size,
                              hipStream_t stream) {
    (void)in_sizes; (void)n_in; (void)out_size; (void)ws_size;
    const float* x      = (const float*)d_in[0];
    const float* W_in   = (const float*)d_in[1];
    const float* conv_w = (const float*)d_in[2];
    const float* conv_b = (const float*)d_in[3];
    const float* W_x    = (const float*)d_in[4];
    const float* W_dt   = (const float*)d_in[5];
    const float* b_dt   = (const float*)d_in[6];
    const float* A_log  = (const float*)d_in[7];
    const float* D_par  = (const float*)d_in[8];
    const float* W_out  = (const float*)d_in[9];
    float* out = (float*)d_out;

    char* ws = (char*)d_ws;
    size_t off = 0;
    auto alloc = [&](size_t bytes) -> char* {
        char* p = ws + off;
        off += (bytes + 255) & ~(size_t)255;
        return p;
    };
    unsigned short* Win_bf  = (unsigned short*)alloc((size_t)2 * MD_INNER * MD_MODEL * 2);
    unsigned short* Wout_bf = (unsigned short*)alloc((size_t)MD_MODEL * MD_INNER * 2);
    unsigned short* Wx_bf   = (unsigned short*)alloc((size_t)XDBL_C * MD_INNER * 2);
    unsigned short* Wdt_bf  = (unsigned short*)alloc((size_t)MD_INNER * MDT_RANK * 2);
    unsigned short* x_bf    = (unsigned short*)alloc((size_t)NTOK * MD_MODEL * 2);
    float*          xr      = (float*)         alloc((size_t)NTOK * 2 * MD_INNER * 4);
    unsigned short* xm_bf   = (unsigned short*)alloc((size_t)NTOK * MD_INNER * 2);
    float*          xdbl    = (float*)         alloc((size_t)NTOK * XDBL_C * 4);
    unsigned short* dlt_bf  = (unsigned short*)alloc((size_t)NTOK * MDT_RANK * 2);
    float*          deltaF  = (float*)         alloc((size_t)NTOK * MD_INNER * 4);
    unsigned short* y_bf    = (unsigned short*)alloc((size_t)NTOK * MD_INNER * 2);

    auto cvt = [&](const float* s, unsigned short* d, int n) {
        k_cvt_bf16<<<2048, 256, 0, stream>>>(s, d, n);
    };
    cvt(W_in,  Win_bf,  2 * MD_INNER * MD_MODEL);
    cvt(W_out, Wout_bf, MD_MODEL * MD_INNER);
    cvt(W_x,   Wx_bf,   XDBL_C * MD_INNER);
    cvt(W_dt,  Wdt_bf,  MD_INNER * MDT_RANK);
    cvt(x,     x_bf,    NTOK * MD_MODEL);

    // big GEMMs: 32x64 block per wave, 8 waves/block
    auto gemm24 = [&](const unsigned short* A, int lda, const unsigned short* W,
                      int ldw, float* C, int ldc, int Mt, int Nt, int K,
                      const float* bias, int ep) {
        int waves = (Mt / 2) * (Nt / 4);
        k_gemm_wmma_2x4<<<(waves + 7) / 8, 256, 0, stream>>>(
            A, lda, W, ldw, C, ldc, Mt / 2, Nt / 4, K, bias, ep);
    };

    // 1) x_and_res = x @ W_in.T
    gemm24(x_bf, MD_MODEL, Win_bf, MD_MODEL, xr, 2 * MD_INNER,
           NTOK / 16, (2 * MD_INNER) / 16, MD_MODEL, nullptr, 0);

    // 2) causal depthwise conv + SiLU
    k_conv_silu<<<dim3(MD_INNER / 256, ML, MB), 256, 0, stream>>>(
        xr, conv_w, conv_b, xm_bf);

    // 3) x_dbl = x_m @ W_x.T  (N=96 -> 6 tiles, single-tile kernel)
    {
        int waves = (NTOK / 16) * (XDBL_C / 16);
        k_gemm_wmma<<<(waves + 7) / 8, 256, 0, stream>>>(
            xm_bf, MD_INNER, Wx_bf, MD_INNER, xdbl, XDBL_C,
            NTOK / 16, XDBL_C / 16, MD_INNER);
    }

    // 4) dlt slice -> bf16
    k_slice_bf16<<<(NTOK * MDT_RANK + 255) / 256, 256, 0, stream>>>(
        xdbl, dlt_bf, NTOK, XDBL_C, MDT_RANK);

    // 5) delta = softplus(dlt @ W_dt.T + b_dt)
    gemm24(dlt_bf, MDT_RANK, Wdt_bf, MDT_RANK, deltaF, MD_INNER,
           NTOK / 16, MD_INNER / 16, MDT_RANK, b_dt, 1);

    // 6) selective scan + skip + gate (fused)
    k_scan_gate<<<dim3(MD_INNER / 256, MB), 256, 0, stream>>>(
        deltaF, xm_bf, xdbl, xr, A_log, D_par, y_bf);

    // 7) out = y @ W_out.T
    gemm24(y_bf, MD_INNER, Wout_bf, MD_INNER, out, MD_MODEL,
           NTOK / 16, MD_MODEL / 16, MD_INNER, nullptr, 0);
}
